// model_36575941492738
// MI455X (gfx1250) — matrix operand order repaired and checked by reading, not compiled, not run on hardware
//
#include <hip/hip_runtime.h>
#include <hip/hip_bf16.h>

// ---------------------------------------------------------------------------
// Sizes from the reference
//   D=64, DIM=256, H=4, N=65, HIDDEN=512, SECTORS=11, B=4096
//   E_DIM = 65*256 + 256*256 + 256 + 256 + 1 = 82689
//   e layout: qw[0:16640], ow[16640:82176], ob[82176:82432], om[82432:82688], outb=e[82688]
// ---------------------------------------------------------------------------

typedef __attribute__((ext_vector_type(16))) _Float16 v16h;
typedef __attribute__((ext_vector_type(8)))  _Float16 v8h;
typedef __attribute__((ext_vector_type(8)))  float    v8f;

#define QSTRIDE 264   // halves per Q row (256 data + pad, 16B-multiple)
#define QROWS   80
#define SSTRIDE 84    // floats per S row (80 data + pad)
#define QBYTES  (QROWS * QSTRIDE * 2)          // 42240
#define SBYTES  (80 * SSTRIDE * 4)             // 26880
#define YOFF    (QBYTES + SBYTES)              // 69120
#define XOFF    (YOFF + 1056)                  // 70176
#define SMEM_C  (XOFF + 256 + 32)              // 70464 bytes dynamic LDS

__device__ __forceinline__ float sigmoidf_(float x) { return 1.0f / (1.0f + __expf(-x)); }

// ---------------------------------------------------------------------------
// Kernel A: batch-independent prefix (single block).
// Computes s_table (11x64), LSTM cell -> h, r = relu(h@W1.T+b1),
// om/ob/outb slices of E, and const = (ob+cls)·om + outb.
// ---------------------------------------------------------------------------
__global__ void __launch_bounds__(256) prep_kernel(
    const float* __restrict__ z,
    const float* __restrict__ emb,  const float* __restrict__ Wsp, const float* __restrict__ bsp,
    const float* __restrict__ Wh,   const float* __restrict__ bh,
    const float* __restrict__ Wc,   const float* __restrict__ bc,
    const float* __restrict__ Wi,   const float* __restrict__ bi,
    const float* __restrict__ W_ih, const float* __restrict__ W_hh,
    const float* __restrict__ b_ih, const float* __restrict__ b_hh,
    const float* __restrict__ W1,   const float* __restrict__ b1,
    const float* __restrict__ W2,   const float* __restrict__ b2,
    const float* __restrict__ cls,
    float* __restrict__ ws_stable, float* __restrict__ ws_r,
    float* __restrict__ ws_om, float* __restrict__ ws_ob, float* __restrict__ ws_const)
{
    __shared__ float h0[64], c0[64], inp[64], gates[256], hv[64], rl[512];
    __shared__ float oml[256], obl[256], outbl;
    const int tid = threadIdx.x;

    // s_table[sec][i] = emb[sec]·Wsp[i] + bsp[i]
    for (int o = tid; o < 11 * 64; o += 256) {
        int sec = o >> 6, i = o & 63;
        float acc = bsp[i];
        for (int j = 0; j < 256; ++j) acc += emb[sec * 256 + j] * Wsp[i * 256 + j];
        ws_stable[o] = acc;
    }

    // h0 / c0 / inp = tanh(z @ W.T + b)
    if (tid < 192) {
        int which = tid >> 6, i = tid & 63;
        const float* W  = (which == 0) ? Wh : (which == 1) ? Wc : Wi;
        const float* bb = (which == 0) ? bh : (which == 1) ? bc : bi;
        float acc = bb[i];
        for (int j = 0; j < 64; ++j) acc += z[j] * W[i * 64 + j];
        float t = tanhf(acc);
        if (which == 0) h0[i] = t; else if (which == 1) c0[i] = t; else inp[i] = t;
    }
    __syncthreads();

    // gates = inp@W_ih.T + b_ih + h0@W_hh.T + b_hh
    {
        float acc = b_ih[tid] + b_hh[tid];
        for (int j = 0; j < 64; ++j)
            acc += inp[j] * W_ih[tid * 64 + j] + h0[j] * W_hh[tid * 64 + j];
        gates[tid] = acc;
    }
    __syncthreads();

    if (tid < 64) {
        float ig = gates[tid], fg = gates[64 + tid], gg = gates[128 + tid], og = gates[192 + tid];
        float cc = sigmoidf_(fg) * c0[tid] + sigmoidf_(ig) * tanhf(gg);
        hv[tid] = sigmoidf_(og) * tanhf(cc);
    }
    __syncthreads();

    // r = relu(h @ W1.T + b1)
    for (int o = tid; o < 512; o += 256) {
        float acc = b1[o];
        for (int j = 0; j < 64; ++j) acc += hv[j] * W1[o * 64 + j];
        acc = fmaxf(acc, 0.0f);
        rl[o] = acc; ws_r[o] = acc;
    }
    __syncthreads();

    // om (256), ob (256), outb : dot(r, W2[k]) + b2[k]
    for (int o = tid; o < 513; o += 256) {
        long k;
        if (o < 256)      k = 82432L + o;          // om
        else if (o < 512) k = 82176L + (o - 256);  // ob
        else              k = 82688L;              // outb
        const float* w = W2 + k * 512;
        float acc = b2[k];
        for (int j = 0; j < 512; ++j) acc += rl[j] * w[j];
        if (o < 256)      { oml[o] = acc;       ws_om[o] = acc; }
        else if (o < 512) { obl[o - 256] = acc; ws_ob[o - 256] = acc; }
        else              outbl = acc;
    }
    __syncthreads();

    if (tid == 0) {
        float acc = outbl;
        for (int j = 0; j < 256; ++j) acc += (obl[j] + cls[j]) * oml[j];
        ws_const[0] = acc;
    }
}

// ---------------------------------------------------------------------------
// Kernel B: grid-parallel slices of E.
//   blocks 0..64   -> qw rows   : qw[m][j] = r·W2[m*256+j] + b2
//   blocks 65..320 -> w_eff[i]  = sum_j (r·W2[16640+i*256+j] + b2) * om[j]
// ---------------------------------------------------------------------------
__global__ void __launch_bounds__(256) edim_kernel(
    const float* __restrict__ W2, const float* __restrict__ b2,
    const float* __restrict__ ws_r, const float* __restrict__ ws_om,
    float* __restrict__ ws_qw, float* __restrict__ ws_weff)
{
    __shared__ float rl[512];
    __shared__ float red[256];
    const int tid = threadIdx.x;
    rl[tid] = ws_r[tid];
    rl[tid + 256] = ws_r[tid + 256];
    __syncthreads();

    const int blk = blockIdx.x;
    if (blk < 65) {
        long k = (long)blk * 256 + tid;
        const float* w = W2 + k * 512;
        float acc = b2[k];
        for (int j = 0; j < 512; ++j) acc += rl[j] * w[j];
        ws_qw[k] = acc;
    } else {
        int i = blk - 65;
        long k = 16640L + (long)i * 256 + tid;
        const float* w = W2 + k * 512;
        float acc = b2[k];
        for (int j = 0; j < 512; ++j) acc += rl[j] * w[j];
        red[tid] = acc * ws_om[tid];
        __syncthreads();
        for (int s = 128; s > 0; s >>= 1) {
            if (tid < s) red[tid] += red[tid + s];
            __syncthreads();
        }
        if (tid == 0) ws_weff[i] = red[0];
    }
}

// ---------------------------------------------------------------------------
// Kernel C: per-batch attention. One block per batch element, 8 waves.
//   Q (65x256) f16 in LDS -> per head S = 0.125 * Q_h Q_h^T via WMMA f16,
//   softmax, write attn, accumulate scalar out via w_eff.
// ---------------------------------------------------------------------------
__global__ void __launch_bounds__(256) attn_kernel(
    const float* __restrict__ X, const int* __restrict__ sector,
    const float* __restrict__ num_w, const float* __restrict__ num_b,
    const float* __restrict__ cls,
    const float* __restrict__ s_table, const float* __restrict__ qw,
    const float* __restrict__ weff, const float* __restrict__ constv,
    float* __restrict__ out_scalar, float* __restrict__ out_attn)
{
    extern __shared__ char smem[];
    _Float16* Qs = (_Float16*)smem;                 // [QROWS][QSTRIDE] f16
    float*    Ss = (float*)(smem + QBYTES);         // [80][SSTRIDE]   f32
    float*    ys = (float*)(smem + YOFF);           // [4*65]
    float*    xs = (float*)(smem + XOFF);           // [64]

    const int b   = blockIdx.x;
    const int tid = threadIdx.x;

    // warm L2-resident shared tables
    __builtin_prefetch(num_w + (size_t)tid * 64, 0, 0);
    __builtin_prefetch(num_b + (size_t)tid * 64, 0, 0);
    __builtin_prefetch(qw    + (size_t)tid * 64, 0, 0);

    const int sec = sector[b];
    if (tid < 64) xs[tid] = X[b * 64 + tid] + s_table[sec * 64 + tid];
    __syncthreads();

    // Build Q = Xt * qw in f16  (row 0 = cls token row)
    for (int idx = tid; idx < 65 * 256; idx += 256) {
        int row = idx >> 8, col = idx & 255;
        float v;
        if (row == 0) {
            v = cls[col] * qw[col];
        } else {
            float xv = xs[row - 1];
            v = (xv * num_w[(row - 1) * 256 + col] + num_b[(row - 1) * 256 + col]) * qw[row * 256 + col];
        }
        Qs[row * QSTRIDE + col] = (_Float16)v;
    }
    __syncthreads();

    // y[h][j] = Q[j, h*64 : h*64+64] · w_eff[h*64 : ...]
    for (int t = tid; t < 4 * 65; t += 256) {
        int h = t / 65, j = t - h * 65;
        float acc = 0.0f;
        for (int k = 0; k < 64; ++k)
            acc += (float)Qs[j * QSTRIDE + h * 64 + k] * weff[h * 64 + k];
        ys[t] = acc;
    }

    const int wave = tid >> 5;
    const int lane = tid & 31;
    const int m16  = lane & 15;
    const int half = lane >> 4;

    float val = 0.0f; // thread 0 only

    for (int h = 0; h < 4; ++h) {
        __syncthreads(); // prior-head S consumers done (and ys ready on h==0)

        // --- S = 0.125 * Q_h Q_h^T, 5x5 tiles of 16x16, K=64 (2 WMMA steps)
        for (int t = wave; t < 25; t += 8) {   // wave-uniform
            const int mi = t / 5, ni = t - mi * 5;
            const int arow = mi * 16 + m16;    // A-matrix row (M)
            const int brow = ni * 16 + m16;    // B = Q^T column -> Q row (N)
            v8f c = {};
            v16h a, bmat;
#pragma unroll
            for (int kk = 0; kk < 2; ++kk) {
                const int kb = h * 64 + kk * 32;
                // A 16x32 f16 layout: lanes0-15 K0-7/K16-23, lanes16-31 K8-15/K24-31
                v8h al = *(const v8h*)(Qs + arow * QSTRIDE + kb + 8 * half);
                v8h ah = *(const v8h*)(Qs + arow * QSTRIDE + kb + 16 + 8 * half);
                // B 32x16 f16 layout: lanes0-15 K0-7/K16-23, lanes16-31 K8-15/K24-31
                v8h bl = *(const v8h*)(Qs + brow * QSTRIDE + kb + 8 * half);
                v8h bh2 = *(const v8h*)(Qs + brow * QSTRIDE + kb + 16 + 8 * half);
#pragma unroll
                for (int e = 0; e < 8; ++e) {
                    a[e] = al[e];  a[e + 8] = ah[e];
                    bmat[e] = bl[e]; bmat[e + 8] = bh2[e];
                }
                c = __builtin_amdgcn_wmma_f32_16x16x32_f16(
                        false, a, false, bmat, (short)0, c, false, false);
            }
            // C/D layout: lane L -> N = L%16 ; VGPR r -> M = r + 8*(L/16)
#pragma unroll
            for (int r = 0; r < 8; ++r) {
                int m = mi * 16 + r + 8 * half;
                int n = ni * 16 + m16;
                Ss[m * SSTRIDE + n] = c[r] * 0.125f;
            }
        }
        __syncthreads();

        // --- softmax over j<65 for rows i<65
        for (int row = tid; row < 65; row += 256) {
            float mx = -1e30f;
            for (int j = 0; j < 65; ++j) mx = fmaxf(mx, Ss[row * SSTRIDE + j]);
            float sum = 0.0f;
            for (int j = 0; j < 65; ++j) {
                float e = __expf(Ss[row * SSTRIDE + j] - mx);
                Ss[row * SSTRIDE + j] = e;
                sum += e;
            }
            float inv = 1.0f / sum;
            for (int j = 0; j < 65; ++j) Ss[row * SSTRIDE + j] *= inv;
        }
        __syncthreads();

        // --- write attn[b][h] (coalesced) + scalar accumulation
        float* dst = out_attn + (size_t)b * (4 * 65 * 65) + (size_t)h * (65 * 65);
        for (int idx = tid; idx < 65 * 65; idx += 256) {
            int i = idx / 65, j = idx - i * 65;
            dst[idx] = Ss[i * SSTRIDE + j];
        }
        if (tid == 0) {
            float acc = 0.0f;
            for (int j = 0; j < 65; ++j) acc += Ss[j] * ys[h * 65 + j];
            val += acc;
        }
    }

    if (tid == 0) out_scalar[b] = val + constv[0];
}

// ---------------------------------------------------------------------------
extern "C" void kernel_launch(void* const* d_in, const int* in_sizes, int n_in,
                              void* d_out, int out_size, void* d_ws, size_t ws_size,
                              hipStream_t stream)
{
    const float* X      = (const float*)d_in[0];
    const float* z      = (const float*)d_in[1];
    const int*   sector = (const int*)  d_in[2];
    const float* emb    = (const float*)d_in[3];
    const float* Wsp    = (const float*)d_in[4];
    const float* bsp    = (const float*)d_in[5];
    const float* Wh     = (const float*)d_in[6];
    const float* bh     = (const float*)d_in[7];
    const float* Wc     = (const float*)d_in[8];
    const float* bc     = (const float*)d_in[9];
    const float* Wi     = (const float*)d_in[10];
    const float* bi     = (const float*)d_in[11];
    const float* W_ih   = (const float*)d_in[12];
    const float* W_hh   = (const float*)d_in[13];
    const float* b_ih   = (const float*)d_in[14];
    const float* b_hh   = (const float*)d_in[15];
    const float* num_w  = (const float*)d_in[16];
    const float* num_b  = (const float*)d_in[17];
    const float* cls    = (const float*)d_in[18];
    const float* W1     = (const float*)d_in[19];
    const float* b1     = (const float*)d_in[20];
    const float* W2     = (const float*)d_in[21];
    const float* b2     = (const float*)d_in[22];

    float* ws        = (float*)d_ws;
    float* ws_stable = ws + 0;      // 704
    float* ws_r      = ws + 704;    // 512
    float* ws_om     = ws + 1216;   // 256
    float* ws_ob     = ws + 1472;   // 256
    float* ws_const  = ws + 1728;   // 1 (padded)
    float* ws_qw     = ws + 1792;   // 16640
    float* ws_weff   = ws + 18432;  // 256

    prep_kernel<<<1, 256, 0, stream>>>(z, emb, Wsp, bsp, Wh, bh, Wc, bc, Wi, bi,
                                       W_ih, W_hh, b_ih, b_hh, W1, b1, W2, b2, cls,
                                       ws_stable, ws_r, ws_om, ws_ob, ws_const);

    edim_kernel<<<321, 256, 0, stream>>>(W2, b2, ws_r, ws_om, ws_qw, ws_weff);

    float* out_scalar = (float*)d_out;
    float* out_attn   = out_scalar + 4096;
    attn_kernel<<<4096, 256, SMEM_C, stream>>>(X, sector, num_w, num_b, cls,
                                               ws_stable, ws_qw, ws_weff, ws_const,
                                               out_scalar, out_attn);
}
